// PointCloudNetPersistencePrediction_32461362823729
// MI455X (gfx1250) — matrix operand order repaired and checked by reading, not compiled, not run on hardware
//
#include <hip/hip_runtime.h>
#include <stdint.h>

// ---------------------------------------------------------------------------
// Geometric scattering on point clouds for MI455X (gfx1250).
// Strategy: materialize the 2048x2048 lazy random-walk matrix P per (b,k)
// pair in f16 (128 MB total, fits MI455X's 192 MB L2), then run the two
// 16-step diffusion chains as WMMA GEMMs (v_wmma_f32_16x16x32_f16) against a
// 2048x16 skinny operand staged column-major (rotated) in LDS.
// Workspace use: ~152 MB of d_ws.
// ---------------------------------------------------------------------------

typedef _Float16 v8h  __attribute__((ext_vector_type(8)));
typedef _Float16 v16h __attribute__((ext_vector_type(16)));
typedef float    v8f  __attribute__((ext_vector_type(8)));

#define NPTS  2048
#define NPAIR 16      // B(4) * K(4)
#define NC    16      // padded feature columns (3 used in bank1, 15 in bank2)
#define SLOT  ((size_t)NPAIR * NPTS * NC)   // elements per saved-level slot

// ---------------------------------------------------------------------------
// Kernel 1: P = 0.5*(I + W/deg), W = thresholded exp(-dist/10), stored f16.
// grid = (2048/256, 16 pairs), block = 256; one row of P per thread.
// ---------------------------------------------------------------------------
__global__ void __launch_bounds__(256)
build_P(const float* __restrict__ pc, const float* __restrict__ al,
        _Float16* __restrict__ P) {
  __shared__ float Xx[NPTS], Xy[NPTS], Xz[NPTS];
  const int pair = blockIdx.y;
  const int b = pair >> 2, k = pair & 3;
  const float a0 = al[k * 3 + 0], a1 = al[k * 3 + 1], a2 = al[k * 3 + 2];
  const float* xb = pc + (size_t)b * NPTS * 3;
  for (int i = threadIdx.x; i < NPTS; i += 256) {
    Xx[i] = xb[i * 3 + 0] * a0;
    Xy[i] = xb[i * 3 + 1] * a1;
    Xz[i] = xb[i * 3 + 2] * a2;
  }
  __syncthreads();

  const int row = blockIdx.x * 256 + threadIdx.x;
  const float x0 = Xx[row], y0 = Xy[row], z0 = Xz[row];

  // pass 1: row degree (diagonal w = exp(0) = 1 is included, so deg >= 1)
  float deg = 0.f;
  for (int j = 0; j < NPTS; ++j) {
    const float dx = x0 - Xx[j], dy = y0 - Xy[j], dz = z0 - Xz[j];
    float w = __expf(-(dx * dx + dy * dy + dz * dz) * 0.1f);
    deg += (w < 0.1f) ? 0.f : w;
  }
  const float inv = 0.5f / deg;

  // pass 2: write P row in f16
  _Float16* prow = P + ((size_t)pair * NPTS + row) * NPTS;
  for (int j = 0; j < NPTS; ++j) {
    const float dx = x0 - Xx[j], dy = y0 - Xy[j], dz = z0 - Xz[j];
    float w = __expf(-(dx * dx + dy * dy + dz * dz) * 0.1f);
    w = (w < 0.1f) ? 0.f : w;
    prow[j] = (_Float16)(w * inv + ((j == row) ? 0.5f : 0.f));
  }
}

// ---------------------------------------------------------------------------
// Kernel 2: F = X * alpha, padded to 16 cols; also stored f32 as level-0.
// ---------------------------------------------------------------------------
__global__ void __launch_bounds__(256)
init_F(const float* __restrict__ pc, const float* __restrict__ al,
       _Float16* __restrict__ F, float* __restrict__ s1_0) {
  const int pair = blockIdx.y;
  const int r = blockIdx.x * 256 + threadIdx.x;
  const int b = pair >> 2, k = pair & 3;
  const size_t base = ((size_t)pair * NPTS + r) * NC;
  for (int c = 0; c < NC; ++c) {
    const float v =
        (c < 3) ? pc[((size_t)b * NPTS + r) * 3 + c] * al[k * 3 + c] : 0.f;
    F[base + c] = (_Float16)v;
    s1_0[base + c] = v;
  }
}

// ---------------------------------------------------------------------------
// Kernel 3: one diffusion step  out = P @ in  (2048x2048 f16 @ 2048x16 f16)
// via v_wmma_f32_16x16x32_f16. grid=(8 row-blocks, 16 pairs), block=256
// (8 waves, each wave owns two 16-row tiles). The skinny operand is staged
// in LDS column-major with a per-column 16-byte rotation (bank-conflict
// free), so the B fragment is two ds_load_b128 per k-chunk.
// D layout follows CDNA5 ISA §7.12.2; A/B use contiguous K = 16h + s.
// ---------------------------------------------------------------------------
__global__ void __launch_bounds__(256)
diffuse_step(const _Float16* __restrict__ P, const _Float16* __restrict__ inY,
             _Float16* __restrict__ outY, float* __restrict__ save) {
  __shared__ _Float16 Ys[NC * NPTS];  // 64 KB, col-major rotated
  const int pair = blockIdx.y;
  const int tid = threadIdx.x;

  // stage inY [2048][16] f16 row-major -> LDS column-major rotated
  {
    const uint32_t* src = (const uint32_t*)(inY + (size_t)pair * NPTS * NC);
    for (int i = tid; i < NPTS * NC / 2; i += 256) {
      union { uint32_t u; _Float16 h[2]; } cv;
      cv.u = src[i];
      const int r = i >> 3;
      const int c = (i & 7) << 1;
      Ys[c * NPTS + ((r + c * 8) & (NPTS - 1))] = cv.h[0];
      Ys[(c + 1) * NPTS + ((r + (c + 1) * 8) & (NPTS - 1))] = cv.h[1];
    }
  }
  __syncthreads();

  const int wave = tid >> 5;
  const int lane = tid & 31;
  const int m = lane & 15;   // A row-in-tile == B column == D column
  const int h = lane >> 4;   // K-half selector
  const int rowbase = blockIdx.x * 256 + wave * 32;

  const _Float16* pr0 = P + ((size_t)pair * NPTS + rowbase + m) * NPTS;
  const _Float16* pr1 = pr0 + (size_t)16 * NPTS;

  v8f c0 = {};
  v8f c1 = {};
  for (int k0 = 0; k0 < NPTS; k0 += 32) {
    // A fragments: lane holds 16 contiguous K at K=16h
    const v8h a0lo = *(const v8h*)(pr0 + k0 + 16 * h);
    const v8h a0hi = *(const v8h*)(pr0 + k0 + 16 * h + 8);
    const v8h a1lo = *(const v8h*)(pr1 + k0 + 16 * h);
    const v8h a1hi = *(const v8h*)(pr1 + k0 + 16 * h + 8);
    // B fragment: lane holds 16 contiguous K at K=16h for column m
    const int xlo = (k0 + 16 * h + m * 8) & (NPTS - 1);
    const int xhi = (xlo + 8) & (NPTS - 1);
    const v8h blo = *(const v8h*)(&Ys[m * NPTS + xlo]);
    const v8h bhi = *(const v8h*)(&Ys[m * NPTS + xhi]);

    const v16h a0 = __builtin_shufflevector(a0lo, a0hi, 0, 1, 2, 3, 4, 5, 6, 7,
                                            8, 9, 10, 11, 12, 13, 14, 15);
    const v16h a1 = __builtin_shufflevector(a1lo, a1hi, 0, 1, 2, 3, 4, 5, 6, 7,
                                            8, 9, 10, 11, 12, 13, 14, 15);
    const v16h bb = __builtin_shufflevector(blo, bhi, 0, 1, 2, 3, 4, 5, 6, 7,
                                            8, 9, 10, 11, 12, 13, 14, 15);

    c0 = __builtin_amdgcn_wmma_f32_16x16x32_f16(false, a0, false, bb,
                                                (short)0, c0, false, false);
    c1 = __builtin_amdgcn_wmma_f32_16x16x32_f16(false, a1, false, bb,
                                                (short)0, c1, false, false);
  }

  // writeback: lane L VGPR v -> (M = v + 8h, N = m)
  _Float16* orow = outY + (size_t)pair * NPTS * NC;
#pragma unroll
  for (int v = 0; v < 8; ++v) {
    const int r0 = rowbase + v + 8 * h;
    orow[(size_t)r0 * NC + m] = (_Float16)c0[v];
    orow[(size_t)(r0 + 16) * NC + m] = (_Float16)c1[v];
  }
  if (save) {
    float* srow = save + (size_t)pair * NPTS * NC;
#pragma unroll
    for (int v = 0; v < 8; ++v) {
      const int r0 = rowbase + v + 8 * h;
      srow[(size_t)r0 * NC + m] = c0[v];
      srow[(size_t)(r0 + 16) * NC + m] = c1[v];
    }
  }
}

// ---------------------------------------------------------------------------
// Kernel 4: first-order wavelets -> U = |L_j - L_{j+1}| packed to 15 cols f16
// ---------------------------------------------------------------------------
__global__ void __launch_bounds__(256)
wavelet1(const float* __restrict__ s1, _Float16* __restrict__ U) {
  const int pair = blockIdx.y;
  const int r = blockIdx.x * 256 + threadIdx.x;
  const size_t base = ((size_t)pair * NPTS + r) * NC;
  _Float16* u = U + base;
  for (int j = 0; j < 5; ++j)
    for (int dd = 0; dd < 3; ++dd)
      u[j * 3 + dd] = (_Float16)fabsf(s1[j * SLOT + base + dd] -
                                      s1[(j + 1) * SLOT + base + dd]);
  u[15] = (_Float16)0.f;
}

// ---------------------------------------------------------------------------
// Kernel 5: per-pair mean over N of the 48 features -> out[pair*48 + f]
// features = [low(3) | firsts(15) | seconds(3+6+9+12)]
// ---------------------------------------------------------------------------
__global__ void __launch_bounds__(256)
finalize(const float* __restrict__ s1, const float* __restrict__ s2,
         float* __restrict__ out) {
  const int pair = blockIdx.x;
  const int tid = threadIdx.x;
  float acc[48];
#pragma unroll
  for (int f = 0; f < 48; ++f) acc[f] = 0.f;

  for (int r = tid; r < NPTS; r += 256) {
    const size_t base = ((size_t)pair * NPTS + r) * NC;
    for (int dd = 0; dd < 3; ++dd) acc[dd] += s1[5 * SLOT + base + dd];
    for (int j = 0; j < 5; ++j)
      for (int dd = 0; dd < 3; ++dd)
        acc[3 + j * 3 + dd] += fabsf(s1[j * SLOT + base + dd] -
                                     s1[(j + 1) * SLOT + base + dd]);
    int o = 18;
    for (int j2 = 1; j2 <= 4; ++j2) {
      for (int c = 0; c < j2 * 3; ++c)
        acc[o + c] += fabsf(s2[(size_t)(j2 - 1) * SLOT + base + c] -
                            s2[(size_t)j2 * SLOT + base + c]);
      o += j2 * 3;
    }
  }

  __shared__ float red[256];
  for (int f = 0; f < 48; ++f) {
    red[tid] = acc[f];
    __syncthreads();
    for (int s = 128; s > 0; s >>= 1) {
      if (tid < s) red[tid] += red[tid + s];
      __syncthreads();
    }
    if (tid == 0) out[pair * 48 + f] = red[0] * (1.0f / NPTS);
    __syncthreads();
  }
}

// ---------------------------------------------------------------------------
extern "C" void kernel_launch(void* const* d_in, const int* in_sizes, int n_in,
                              void* d_out, int out_size, void* d_ws,
                              size_t ws_size, hipStream_t stream) {
  (void)in_sizes; (void)n_in; (void)out_size; (void)ws_size;
  const float* pc = (const float*)d_in[0];  // [4][2048][3] f32
  const float* al = (const float*)d_in[1];  // [4][3] f32
  float* out = (float*)d_out;               // [4][192] f32

  char* ws = (char*)d_ws;
  _Float16* P = (_Float16*)ws;  // 128 MB
  size_t off = (size_t)NPAIR * NPTS * NPTS * sizeof(_Float16);
  const size_t ybytes = (size_t)NPAIR * NPTS * NC * sizeof(_Float16);
  _Float16* curA = (_Float16*)(ws + off); off += ybytes;
  _Float16* curB = (_Float16*)(ws + off); off += ybytes;
  const size_t sbytes = SLOT * sizeof(float);
  float* s1 = (float*)(ws + off); off += 6 * sbytes;  // levels t=0,1,2,4,8,16
  float* s2 = (float*)(ws + off); off += 5 * sbytes;  // levels t=1,2,4,8,16

  const dim3 g8(NPTS / 256, NPAIR);

  build_P<<<g8, 256, 0, stream>>>(pc, al, P);
  init_F<<<g8, 256, 0, stream>>>(pc, al, curA, s1 /* slot 0 */);

  _Float16* in = curA;
  _Float16* outp = curB;
  for (int t = 1; t <= 16; ++t) {
    float* sv = nullptr;
    if (t == 1) sv = s1 + 1 * SLOT;
    else if (t == 2) sv = s1 + 2 * SLOT;
    else if (t == 4) sv = s1 + 3 * SLOT;
    else if (t == 8) sv = s1 + 4 * SLOT;
    else if (t == 16) sv = s1 + 5 * SLOT;
    diffuse_step<<<g8, 256, 0, stream>>>(P, in, outp, sv);
    _Float16* t2 = in; in = outp; outp = t2;
  }

  // 16 steps -> `in` is back to curA and is free; write U there.
  wavelet1<<<g8, 256, 0, stream>>>(s1, in);

  for (int t = 1; t <= 16; ++t) {
    float* sv = nullptr;
    if (t == 1) sv = s2 + 0 * SLOT;
    else if (t == 2) sv = s2 + 1 * SLOT;
    else if (t == 4) sv = s2 + 2 * SLOT;
    else if (t == 8) sv = s2 + 3 * SLOT;
    else if (t == 16) sv = s2 + 4 * SLOT;
    diffuse_step<<<g8, 256, 0, stream>>>(P, in, outp, sv);
    _Float16* t2 = in; in = outp; outp = t2;
  }

  finalize<<<NPAIR, 256, 0, stream>>>(s1, s2, out);
}
